// DepthRend_50199577755651
// MI455X (gfx1250) — matrix operand order repaired and checked by reading, not compiled, not run on hardware
//
#include <hip/hip_runtime.h>
#include <cstdint>

// ---------------------------------------------------------------------------
// Problem constants (from reference setup_inputs)
//   B=4, C=90, CF=2560, HID=512, H0=48, W0=64, scales: 96x128/192x256/384x512
// ---------------------------------------------------------------------------
#define BB    4
#define CC    90
#define CF    2560
#define HID   512
#define HF    48
#define WF    64
#define NMAX  4096
#define KW    64          // K elements staged per LDS buffer (2 WMMA K-steps)
#define PITCH 72          // LDS row pitch in halfs: 64 data + 8 pad -> 144B
                          // (bank stride 36 mod 64 -> conflict-free b128 reads)

typedef _Float16 h16;
typedef __attribute__((ext_vector_type(16))) _Float16 v16h;
typedef __attribute__((ext_vector_type(8)))  _Float16 v8h;
typedef __attribute__((ext_vector_type(8)))  float    v8f;

typedef unsigned int u32x4 __attribute__((ext_vector_type(4)));
typedef int          i32x8 __attribute__((ext_vector_type(8)));
typedef int          i32x4 __attribute__((ext_vector_type(4)));

#if defined(__has_builtin)
#if __has_builtin(__builtin_amdgcn_tensor_load_to_lds) && __has_builtin(__builtin_amdgcn_s_wait_tensorcnt)
#define USE_TDM 1
#endif
#endif
#ifndef USE_TDM
#define USE_TDM 0
#endif

#if USE_TDM
// 2-D tile (rows x 64 halfs) DMA: global -> LDS via Tensor Data Mover.
// LDS padding: after every 32 DWORDs (one 128B row) insert 4 DWORDs -> 144B pitch.
__device__ __forceinline__ void tdm_load_tile(uint32_t ldsByteOff, const void* g,
                                              int rows, int rowStrideHalfs) {
  uint64_t ga = (uint64_t)(uintptr_t)g;
  u32x4 g0;
  g0[0] = 1u;                                   // count = 1 (valid descriptor)
  g0[1] = ldsByteOff;                           // lds_addr
  g0[2] = (uint32_t)ga;                         // global_addr[31:0]
  g0[3] = (uint32_t)(ga >> 32) | (2u << 30);    // global_addr[56:32] | type=2
  i32x8 g1;
  g1[0] = (int)((1u << 16)                      // data_size = 2 bytes
              | (1u << 20)                      // pad_enable
              | (4u << 22)                      // pad_interval: 32 DWORDs
              | (3u << 25));                    // pad_amount: 4 DWORDs
  g1[1] = (int)((unsigned)KW << 16);            // tensor_dim0 = 64
  g1[2] = (int)(((uint32_t)rows & 0xFFFFu) << 16); // tensor_dim1 = rows
  g1[3] = (int)((unsigned)KW << 16);            // tile_dim0 = 64
  g1[4] = rows & 0xFFFF;                        // tile_dim1 = rows, tile_dim2 = 0
  g1[5] = rowStrideHalfs;                       // tensor_dim0_stride (elements)
  g1[6] = 0;
  g1[7] = 0;
  i32x4 z = {0, 0, 0, 0};
#if __clang_major__ >= 23
  i32x8 z8 = {0, 0, 0, 0, 0, 0, 0, 0};
  __builtin_amdgcn_tensor_load_to_lds(g0, g1, z, z, z8, 0);
#else
  __builtin_amdgcn_tensor_load_to_lds(g0, g1, z, z, 0);
#endif
}
#endif

// Cooperative fallback loader (used only if TDM builtin is unavailable)
__device__ __forceinline__ void coop_load(h16* lds, const h16* g, int rows,
                                          int strideHalfs, int tid) {
  for (int i = tid; i < rows * 8; i += 256) {
    int r = i >> 3, c = (i & 7) << 3;
    *(v8h*)(lds + r * PITCH + c) = *(const v8h*)(g + (size_t)r * strideHalfs + c);
  }
}

__device__ __forceinline__ v16h mk16(v8h lo, v8h hi) {
  v16h r;
#pragma unroll
  for (int i = 0; i < 8; ++i) { r[i] = lo[i]; r[i + 8] = hi[i]; }
  return r;
}

// A operand (16x32 f16): lanes 0-15 carry K 0..7 & 16..23, lanes 16-31 carry 8..15 & 24..31
__device__ __forceinline__ v16h fragA(const h16* lds, int rowBase, int kBase, int lane) {
  int r = rowBase + (lane & 15);
  int k0 = (lane & 16) ? 8 : 0;
  const h16* p = lds + r * PITCH + kBase;
  return mk16(*(const v8h*)(p + k0), *(const v8h*)(p + k0 + 16));
}

// B operand (32x16 f16), tile stored N-major: lanes 0-15 carry K 0..7 & 16..23, lanes 16-31 carry 8..15 & 24..31
__device__ __forceinline__ v16h fragB(const h16* lds, int rowBase, int kBase, int lane) {
  int r = rowBase + (lane & 15);
  int k0 = (lane & 16) ? 8 : 0;
  const h16* p = lds + r * PITCH + kBase;
  return mk16(*(const v8h*)(p + k0), *(const v8h*)(p + k0 + 16));
}

#define WMMA16(A, Bf, Cacc) \
  __builtin_amdgcn_wmma_f32_16x16x32_f16(false, (A), false, (Bf), (short)0, (Cacc), false, false)

// ---------------------------------------------------------------------------
// GEMM1: hdn[n,512] = relu(fpts[n,2560] @ W1 + b1), f16 WMMA, f16 output
// block = 256 threads (8 waves), 128x64 output tile, double-buffered TDM,
// K staged 64 deep -> 8 WMMAs per barrier pair, DMA overlapped with compute.
// ---------------------------------------------------------------------------
__global__ __launch_bounds__(256) void dr_gemm1(const h16* __restrict__ fpts,
                                                const h16* __restrict__ w1t,
                                                const float* __restrict__ bias1,
                                                h16* __restrict__ hdn, int npts) {
  __shared__ __align__(16) h16 lA[2][128 * PITCH];
  __shared__ __align__(16) h16 lB[2][64 * PITCH];
  const int b = blockIdx.z, mBase = blockIdx.x * 128, nBase = blockIdx.y * 64;
  const int tid = threadIdx.x, lane = tid & 31, wave = tid >> 5;
  const h16* Ag = fpts + ((size_t)b * npts + mBase) * CF;
  const h16* Bg = w1t + (size_t)nBase * CF;
  const int mT0 = (wave & 3) * 2, nT0 = (wave >> 2) * 2;
  v8f acc00 = {0,0,0,0,0,0,0,0}, acc01 = {0,0,0,0,0,0,0,0};
  v8f acc10 = {0,0,0,0,0,0,0,0}, acc11 = {0,0,0,0,0,0,0,0};
  const int KT = CF / KW;   // 40 stages
#if USE_TDM
  const uint32_t offA[2] = {(uint32_t)(uintptr_t)&lA[0][0], (uint32_t)(uintptr_t)&lA[1][0]};
  const uint32_t offB[2] = {(uint32_t)(uintptr_t)&lB[0][0], (uint32_t)(uintptr_t)&lB[1][0]};
  if (wave == 0) {                       // prologue: stage 0 into buffer 0
    tdm_load_tile(offA[0], Ag, 128, CF);
    tdm_load_tile(offB[0], Bg, 64, CF);
  }
#endif
  for (int kt = 0; kt < KT; ++kt) {
    const int cur = kt & 1;
    __syncthreads();                     // buffer (cur^1) fully consumed
#if USE_TDM
    if (wave == 0) {
      if (kt + 1 < KT) {
        tdm_load_tile(offA[cur ^ 1], Ag + (size_t)(kt + 1) * KW, 128, CF);
        tdm_load_tile(offB[cur ^ 1], Bg + (size_t)(kt + 1) * KW, 64, CF);
        __builtin_amdgcn_s_wait_tensorcnt(2);  // stage kt complete (in-order)
      } else {
        __builtin_amdgcn_s_wait_tensorcnt(0);
      }
    }
#else
    coop_load(lA[cur], Ag + (size_t)kt * KW, 128, CF, tid);
    coop_load(lB[cur], Bg + (size_t)kt * KW, 64, CF, tid);
#endif
    __syncthreads();                     // publish buffer cur
    const h16* A = lA[cur];
    const h16* Bp = lB[cur];
#pragma unroll
    for (int s = 0; s < 2; ++s) {
      v16h a0 = fragA(A, mT0 * 16, s * 32, lane);
      v16h a1 = fragA(A, (mT0 + 1) * 16, s * 32, lane);
      v16h b0 = fragB(Bp, nT0 * 16, s * 32, lane);
      v16h b1 = fragB(Bp, (nT0 + 1) * 16, s * 32, lane);
      acc00 = WMMA16(a0, b0, acc00);
      acc01 = WMMA16(a0, b1, acc01);
      acc10 = WMMA16(a1, b0, acc10);
      acc11 = WMMA16(a1, b1, acc11);
    }
  }
  const int mHi = (lane & 16) ? 8 : 0;
  const int mRow0 = mBase + mT0 * 16 + mHi;
  const int mRow1 = mBase + (mT0 + 1) * 16 + mHi;
  const int col0 = nBase + nT0 * 16 + (lane & 15);
  const int col1 = col0 + 16;
  const float bb0 = bias1[col0], bb1 = bias1[col1];
#pragma unroll
  for (int r = 0; r < 8; ++r) {
    const size_t m0 = (size_t)b * npts + mRow0 + r;
    const size_t m1 = (size_t)b * npts + mRow1 + r;
    float v;
    v = acc00[r] + bb0; hdn[m0 * HID + col0] = (h16)(v > 0.f ? v : 0.f);
    v = acc01[r] + bb1; hdn[m0 * HID + col1] = (h16)(v > 0.f ? v : 0.f);
    v = acc10[r] + bb0; hdn[m1 * HID + col0] = (h16)(v > 0.f ? v : 0.f);
    v = acc11[r] + bb1; hdn[m1 * HID + col1] = (h16)(v > 0.f ? v : 0.f);
  }
}

// ---------------------------------------------------------------------------
// GEMM2: outm[n,96] = hdn[n,512] @ W2 + b2 (cols 90..95 padded), f32 output
// block = 128x96 output tile, double-buffered TDM, 12 WMMAs per barrier pair
// ---------------------------------------------------------------------------
__global__ __launch_bounds__(256) void dr_gemm2(const h16* __restrict__ hdn,
                                                const h16* __restrict__ w2t,
                                                const float* __restrict__ bias2,
                                                float* __restrict__ outm, int npts) {
  __shared__ __align__(16) h16 lA[2][128 * PITCH];
  __shared__ __align__(16) h16 lB[2][96 * PITCH];
  const int b = blockIdx.z, mBase = blockIdx.x * 128;
  const int tid = threadIdx.x, lane = tid & 31, wave = tid >> 5;
  const h16* Ag = hdn + ((size_t)b * npts + mBase) * HID;
  const h16* Bg = w2t;
  const int mT0 = (wave & 3) * 2, nT0 = (wave >> 2) * 3;
  v8f zero = {0,0,0,0,0,0,0,0};
  v8f acc[2][3] = {{zero, zero, zero}, {zero, zero, zero}};
  const int KT = HID / KW;   // 8 stages
#if USE_TDM
  const uint32_t offA[2] = {(uint32_t)(uintptr_t)&lA[0][0], (uint32_t)(uintptr_t)&lA[1][0]};
  const uint32_t offB[2] = {(uint32_t)(uintptr_t)&lB[0][0], (uint32_t)(uintptr_t)&lB[1][0]};
  if (wave == 0) {
    tdm_load_tile(offA[0], Ag, 128, HID);
    tdm_load_tile(offB[0], Bg, 96, HID);
  }
#endif
  for (int kt = 0; kt < KT; ++kt) {
    const int cur = kt & 1;
    __syncthreads();
#if USE_TDM
    if (wave == 0) {
      if (kt + 1 < KT) {
        tdm_load_tile(offA[cur ^ 1], Ag + (size_t)(kt + 1) * KW, 128, HID);
        tdm_load_tile(offB[cur ^ 1], Bg + (size_t)(kt + 1) * KW, 96, HID);
        __builtin_amdgcn_s_wait_tensorcnt(2);
      } else {
        __builtin_amdgcn_s_wait_tensorcnt(0);
      }
    }
#else
    coop_load(lA[cur], Ag + (size_t)kt * KW, 128, HID, tid);
    coop_load(lB[cur], Bg + (size_t)kt * KW, 96, HID, tid);
#endif
    __syncthreads();
    const h16* A = lA[cur];
    const h16* Bp = lB[cur];
#pragma unroll
    for (int s = 0; s < 2; ++s) {
      v16h a0 = fragA(A, mT0 * 16, s * 32, lane);
      v16h a1 = fragA(A, (mT0 + 1) * 16, s * 32, lane);
#pragma unroll
      for (int j = 0; j < 3; ++j) {
        v16h bf = fragB(Bp, (nT0 + j) * 16, s * 32, lane);
        acc[0][j] = WMMA16(a0, bf, acc[0][j]);
        acc[1][j] = WMMA16(a1, bf, acc[1][j]);
      }
    }
  }
  const int mHi = (lane & 16) ? 8 : 0;
#pragma unroll
  for (int i = 0; i < 2; ++i) {
    const int mRow = mBase + (mT0 + i) * 16 + mHi;
#pragma unroll
    for (int j = 0; j < 3; ++j) {
      const int col = (nT0 + j) * 16 + (lane & 15);
      if (col < CC) {
        const float bb = bias2[col];
#pragma unroll
        for (int r = 0; r < 8; ++r)
          outm[((size_t)b * npts + mRow + r) * 96 + col] = acc[i][j][r] + bb;
      }
    }
  }
}

// ---------------------------------------------------------------------------
// Bilinear 2x upsample (half-pixel centers, edge clamp) over `planes` planes
// ---------------------------------------------------------------------------
__global__ void dr_up2(const float* __restrict__ src, float* __restrict__ dst,
                       int planes, int ih, int iw) {
  const int oh = ih * 2, ow = iw * 2;
  const size_t total = (size_t)planes * oh * ow;
  const size_t o = (size_t)blockIdx.x * blockDim.x + threadIdx.x;
  if (o >= total) return;
  const int ox = (int)(o % ow);
  const size_t t = o / ow;
  const int oy = (int)(t % oh);
  const int pl = (int)(t / oh);
  float ty = fminf(fmaxf((oy + 0.5f) * 0.5f - 0.5f, 0.f), (float)(ih - 1));
  float tx = fminf(fmaxf((ox + 0.5f) * 0.5f - 0.5f, 0.f), (float)(iw - 1));
  int y0 = (int)ty; if (y0 > ih - 2) y0 = ih - 2;
  int x0 = (int)tx; if (x0 > iw - 2) x0 = iw - 2;
  const float wy = ty - y0, wx = tx - x0;
  const float* p = src + (size_t)pl * ih * iw;
  const float f00 = p[y0 * iw + x0],       f01 = p[y0 * iw + x0 + 1];
  const float f10 = p[(y0 + 1) * iw + x0], f11 = p[(y0 + 1) * iw + x0 + 1];
  const float top = f00 + (f01 - f00) * wx;
  const float bot = f10 + (f11 - f10) * wx;
  dst[o] = top + (bot - top) * wy;
}

// ---------------------------------------------------------------------------
// Softmax top2 uncertainty: unc = p2 - p1 (<=0; larger == more uncertain)
// ---------------------------------------------------------------------------
__global__ void dr_unc(const float* __restrict__ logit, float* __restrict__ unc, int hw) {
  const int b = blockIdx.y;
  const int p = blockIdx.x * blockDim.x + threadIdx.x;
  if (p >= hw) return;
  const float* lp = logit + (size_t)b * CC * hw + p;
  float m1 = -3.4e38f, m2 = -3.4e38f, s = 0.f;
  for (int c = 0; c < CC; ++c) {
    const float v = lp[(size_t)c * hw];
    if (v > m1) { s = s * __expf(m1 - v) + 1.f; m2 = m1; m1 = v; }
    else        { s += __expf(v - m1); if (v > m2) m2 = v; }
  }
  unc[(size_t)b * hw + p] = (__expf(m2 - m1) - 1.f) / s;
}

// ---------------------------------------------------------------------------
// Exact top-k via 4-round radix select on order-preserving float keys.
// Per-batch state (272 u32): hist[256], prefix, kneed, thr, pad, kEQ, cGT, cEQ
// ---------------------------------------------------------------------------
__device__ __forceinline__ unsigned ordkey(float f) {
  unsigned k = __float_as_uint(f);
  return (k & 0x80000000u) ? ~k : (k | 0x80000000u);
}

__global__ void dr_tk_init(unsigned* tk, int npts) {
  for (int i = threadIdx.x; i < BB * 272; i += blockDim.x) tk[i] = 0;
  __syncthreads();
  if (threadIdx.x < BB) tk[threadIdx.x * 272 + 257] = (unsigned)npts;
}

__global__ void dr_tk_hist(const float* __restrict__ unc, unsigned* tk, int hw, int r) {
  __shared__ unsigned sh[256];
  const int b = blockIdx.y;
  unsigned* st = tk + (size_t)b * 272;
  sh[threadIdx.x] = 0;
  __syncthreads();
  const unsigned pref = st[256];
  const int selShift = 32 - 8 * r;   // only evaluated when r >= 1
  const int binShift = 24 - 8 * r;
  const float* u = unc + (size_t)b * hw;
  for (int p = blockIdx.x * blockDim.x + threadIdx.x; p < hw; p += gridDim.x * blockDim.x) {
    const unsigned k = ordkey(u[p]);
    const bool ok = (r == 0) || ((k >> selShift) == (pref >> selShift));
    if (ok) atomicAdd(&sh[(k >> binShift) & 255u], 1u);
  }
  __syncthreads();
  if (sh[threadIdx.x]) atomicAdd(&st[threadIdx.x], sh[threadIdx.x]);
}

__global__ void dr_tk_scan(unsigned* tk, int r, int npts) {
  if (threadIdx.x != 0) return;
  unsigned* st = tk + (size_t)blockIdx.x * 272;
  const unsigned kneed = st[257];
  unsigned cum = 0; int bin = 0;
  for (int i = 255; i >= 0; --i) {
    const unsigned c = st[i];
    if (cum + c >= kneed) { bin = i; break; }
    cum += c;
  }
  st[256] |= ((unsigned)bin) << (24 - 8 * r);
  st[257] = kneed - cum;
  for (int i = 0; i < 256; ++i) st[i] = 0;
  if (r == 3) {
    st[258] = st[256];   // threshold key
    st[260] = st[257];   // # of threshold-equal elements needed
    st[261] = 0;         // >thr counter
    st[262] = 0;         // ==thr counter
  }
  (void)npts;
}

__global__ void dr_tk_compact(const float* __restrict__ unc, unsigned* tk,
                              int* __restrict__ idx, int hw, int npts) {
  const int b = blockIdx.y;
  unsigned* st = tk + (size_t)b * 272;
  const unsigned thr = st[258], kEQ = st[260];
  const float* u = unc + (size_t)b * hw;
  for (int p = blockIdx.x * blockDim.x + threadIdx.x; p < hw; p += gridDim.x * blockDim.x) {
    const unsigned k = ordkey(u[p]);
    if (k > thr) {
      const unsigned pos = atomicAdd(&st[261], 1u);
      if (pos < (unsigned)npts) idx[b * NMAX + pos] = p;
    } else if (k == thr) {
      const unsigned q = atomicAdd(&st[262], 1u);
      if (q < kEQ) idx[b * NMAX + (npts - 1 - (int)q)] = p;
    }
  }
}

// ---------------------------------------------------------------------------
// Bilinear gather of 2560-ch feature vectors at selected points -> f16 rows
// ---------------------------------------------------------------------------
__global__ __launch_bounds__(256) void dr_gather(const float* __restrict__ feat,
                                                 const int* __restrict__ idx,
                                                 h16* __restrict__ fpts,
                                                 int npts, int h, int w) {
  const int b = blockIdx.y, pt = blockIdx.x;
  const int pix = idx[b * NMAX + pt];
  const int iy = pix / w, ix = pix - iy * w;
  const float ys = ((iy + 0.5f) / (float)h) * (float)HF - 0.5f;
  const float xs = ((ix + 0.5f) / (float)w) * (float)WF - 0.5f;
  float ty = fminf(fmaxf(ys, 0.f), (float)(HF - 1));
  float tx = fminf(fmaxf(xs, 0.f), (float)(WF - 1));
  int y0 = (int)ty; if (y0 > HF - 2) y0 = HF - 2;
  int x0 = (int)tx; if (x0 > WF - 2) x0 = WF - 2;
  const float wy = ty - y0, wx = tx - x0;
  const float* fb = feat + (size_t)b * CF * (HF * WF);
  const size_t base = (size_t)y0 * WF + x0;
  h16* dst = fpts + ((size_t)b * npts + pt) * CF;
  for (int c = threadIdx.x; c < CF; c += 256) {
    const float* fc = fb + (size_t)c * (HF * WF);
    const float f00 = fc[base], f01 = fc[base + 1];
    const float f10 = fc[base + WF], f11 = fc[base + WF + 1];
    const float top = f00 + (f01 - f00) * wx;
    const float bot = f10 + (f11 - f10) * wx;
    dst[c] = (h16)(top + (bot - top) * wy);
  }
}

// ---------------------------------------------------------------------------
// Scatter MLP outputs into logit; mark painter
// ---------------------------------------------------------------------------
__global__ void dr_scatter(const float* __restrict__ outm, const int* __restrict__ idx,
                           float* __restrict__ logit, float* __restrict__ painter,
                           int npts, int hw) {
  const int b = blockIdx.y, pt = blockIdx.x, c = threadIdx.x;
  const int pix = idx[b * NMAX + pt];
  if (c < CC)
    logit[((size_t)b * CC + c) * hw + pix] = outm[((size_t)b * npts + pt) * 96 + c];
  else if (c == CC)
    painter[(size_t)b * hw + pix] = 1.f;
}

// ---------------------------------------------------------------------------
// Weight prep: W1T[h, c] = W1[c, h] as f16; W2T[c(pad 96), h] = W2[h, c] as f16
// ---------------------------------------------------------------------------
__global__ void dr_w1t(const float* __restrict__ W1, h16* __restrict__ W1T) {
  const int id = blockIdx.x * 256 + threadIdx.x;
  if (id >= HID * CF) return;
  const int c = id % CF, hcol = id / CF;
  W1T[id] = (h16)W1[(size_t)c * HID + hcol];
}

__global__ void dr_w2t(const float* __restrict__ W2, h16* __restrict__ W2T) {
  const int id = blockIdx.x * 256 + threadIdx.x;
  if (id >= 96 * HID) return;
  const int hcol = id % HID, c = id / HID;
  W2T[id] = (c < CC) ? (h16)W2[(size_t)hcol * CC + c] : (h16)0.f;
}

__global__ void dr_zero(float* p, int n) {
  const int i = blockIdx.x * 256 + threadIdx.x;
  if (i < n) p[i] = 0.f;
}

// ---------------------------------------------------------------------------
// Host orchestration (all on `stream`, graph-capture safe, ws-only scratch)
// ws budget ~203 MB.
// ---------------------------------------------------------------------------
extern "C" void kernel_launch(void* const* d_in, const int* in_sizes, int n_in,
                              void* d_out, int out_size, void* d_ws, size_t ws_size,
                              hipStream_t stream) {
  (void)in_sizes; (void)n_in; (void)out_size; (void)ws_size;
  const float* logit_in = (const float*)d_in[0];
  const float* feature  = (const float*)d_in[1];
  const float* W1 = (const float*)d_in[2];
  const float* b1 = (const float*)d_in[3];
  const float* W2 = (const float*)d_in[4];
  const float* b2 = (const float*)d_in[5];

  float* outLogit   = (float*)d_out;
  float* outPainter = outLogit + (size_t)BB * CC * 384 * 512;

  char* ws = (char*)d_ws;
  size_t off = 0;
  auto take = [&](size_t bytes) -> char* {
    char* p = ws + off;
    off = (off + bytes + 255) & ~(size_t)255;
    return p;
  };
  float*    L1   = (float*)take((size_t)BB * CC * 96 * 128 * 4);
  float*    L2   = (float*)take((size_t)BB * CC * 192 * 256 * 4);
  float*    pA   = (float*)take((size_t)BB * 96 * 128 * 4);
  float*    pB   = (float*)take((size_t)BB * 192 * 256 * 4);
  float*    unc  = (float*)take((size_t)BB * 384 * 512 * 4);
  int*      idx  = (int*)take((size_t)BB * NMAX * 4);
  unsigned* tk   = (unsigned*)take((size_t)BB * 272 * 4);
  h16*      fpts = (h16*)take((size_t)BB * NMAX * CF * 2);
  h16*      hdn  = (h16*)take((size_t)BB * NMAX * HID * 2);
  float*    outm = (float*)take((size_t)BB * NMAX * 96 * 4);
  h16*      W1T  = (h16*)take((size_t)HID * CF * 2);
  h16*      W2T  = (h16*)take((size_t)96 * HID * 2);

  dr_w1t<<<(HID * CF + 255) / 256, 256, 0, stream>>>(W1, W1T);
  dr_w2t<<<(96 * HID + 255) / 256, 256, 0, stream>>>(W2, W2T);
  dr_zero<<<(BB * 96 * 128 + 255) / 256, 256, 0, stream>>>(pA, BB * 96 * 128);

  struct Sc { int h, w, npts; const float* lsrc; float* ldst; const float* psrc; float* pdst; };
  const Sc sc[3] = {
    { 96, 128,  256, logit_in, L1,       nullptr, pA         },
    {192, 256, 1024, L1,       L2,       pA,      pB         },
    {384, 512, 4096, L2,       outLogit, pB,      outPainter },
  };

  for (int i = 0; i < 3; ++i) {
    const Sc s = sc[i];
    const int hw = s.h * s.w;

    {
      const int total = BB * CC * hw;
      dr_up2<<<(total + 255) / 256, 256, 0, stream>>>(s.lsrc, s.ldst, BB * CC, s.h / 2, s.w / 2);
    }
    if (s.psrc) {
      const int total = BB * hw;
      dr_up2<<<(total + 255) / 256, 256, 0, stream>>>(s.psrc, s.pdst, BB, s.h / 2, s.w / 2);
    }

    dr_unc<<<dim3((hw + 255) / 256, BB), 256, 0, stream>>>(s.ldst, unc, hw);

    dr_tk_init<<<1, 256, 0, stream>>>(tk, s.npts);
    for (int r = 0; r < 4; ++r) {
      dr_tk_hist<<<dim3(192, BB), 256, 0, stream>>>(unc, tk, hw, r);
      dr_tk_scan<<<BB, 32, 0, stream>>>(tk, r, s.npts);
    }
    dr_tk_compact<<<dim3(192, BB), 256, 0, stream>>>(unc, tk, idx, hw, s.npts);

    dr_gather<<<dim3(s.npts, BB), 256, 0, stream>>>(feature, idx, fpts, s.npts, s.h, s.w);

    dr_gemm1<<<dim3(s.npts / 128, HID / 64, BB), 256, 0, stream>>>(fpts, W1T, b1, hdn, s.npts);
    dr_gemm2<<<dim3(s.npts / 128, 1, BB), 256, 0, stream>>>(hdn, W2T, b2, outm, s.npts);

    dr_scatter<<<dim3(s.npts, BB), 96, 0, stream>>>(outm, idx, s.ldst, s.pdst, s.npts, hw);
  }
}
